// SpikingGroupedSlidingAttention_1563368095958
// MI455X (gfx1250) — compile-verified, matrix operand order checked by reading, not run on hardware
//
#include <hip/hip_runtime.h>
#include <hip/hip_bf16.h>

// ---------------------------------------------------------------------------
// Types for CDNA5 WMMA (gfx1250, wave32)
// ---------------------------------------------------------------------------
typedef __bf16 bf16_t;
typedef __attribute__((ext_vector_type(16))) __bf16 v16bf;
typedef __attribute__((ext_vector_type(8)))  __bf16 v8bf;
typedef __attribute__((ext_vector_type(8)))  float  v8f;

#define B_DIM      2
#define S_DIM      1024
#define D_MODEL    2048
#define N_HEADS    16
#define N_KV       4
#define HEAD_DIM   128
#define WINDOW     128
#define GROUPS     (N_HEADS / N_KV)   // 4
#define SM_SCALE   0.08838834764831845f  // 1/sqrt(128)

static __device__ __forceinline__ v8f wmma_bf16(v16bf a, v16bf b, v8f c) {
  // D = A(16x32 bf16) * B(32x16 bf16) + C(16x16 f32)
  return __builtin_amdgcn_wmma_f32_16x16x32_bf16(
      /*neg_a=*/false, a, /*neg_b=*/false, b,
      /*c_mod=*/(short)0, c, /*reuse_a=*/false, /*reuse_b=*/false);
}

// Build a 16-element bf16 fragment from two contiguous 16B chunks.
static __device__ __forceinline__ v16bf frag_from(const bf16_t* p) {
  v8bf lo = *(const v8bf*)(p);
  v8bf hi = *(const v8bf*)(p + 16);
  return __builtin_shufflevector(lo, hi,
      0,1,2,3,4,5,6,7,8,9,10,11,12,13,14,15);
}

// A/B fragment loader for 16-bit WMMA operands (ISA 16x32 layout):
//   lane L -> row/col (L & 15)
//   elements i=0..7  -> k = (L>>4)*8 + i
//   elements i=8..15 -> k = 16 + (L>>4)*8 + (i-8)
// `base` points at (row0, k0) of a row-major [rows, ld] bf16 array.
static __device__ __forceinline__ v16bf load_frag(const bf16_t* base, int ld) {
  int lane = threadIdx.x & 31;
  const bf16_t* p = base + (size_t)(lane & 15) * ld + (lane >> 4) * 8;
  return frag_from(p);
}

// ---------------------------------------------------------------------------
// Kernel 1: f32 -> bf16 elementwise convert (for x)
// ---------------------------------------------------------------------------
__global__ void k_convert(const float* __restrict__ src,
                          bf16_t* __restrict__ dst, int n) {
  int i = blockIdx.x * blockDim.x + threadIdx.x;
  if (i < n) dst[i] = (bf16_t)src[i];
}

// ---------------------------------------------------------------------------
// Kernel 2: W[K,N] f32 -> Wt[N,K] bf16 (transpose+convert) so B-fragments
// are contiguous along K.
// ---------------------------------------------------------------------------
__global__ void k_transpose_w(const float* __restrict__ W,
                              bf16_t* __restrict__ Wt, int K, int N) {
  int i = blockIdx.x * blockDim.x + threadIdx.x;
  if (i >= K * N) return;
  int k = i / N;
  int n = i - k * N;
  Wt[(size_t)n * K + k] = (bf16_t)W[i];
}

// ---------------------------------------------------------------------------
// Kernel 3: generic bf16 WMMA GEMM  C[M,N] = A[M,K] * Bt[N,K]^T  (bf16 out)
//   block = 128 threads (4 waves); wave computes a 16x64 tile (4 N-subtiles)
//   grid  = (M/16, N/256)
//   Software-pipelined: fragments for k-step i+1 are loaded before the
//   WMMAs of step i issue, so global/L2 latency is hidden behind 4 WMMAs.
// ---------------------------------------------------------------------------
__global__ void k_gemm_bf16(const bf16_t* __restrict__ A,
                            const bf16_t* __restrict__ Bt,
                            bf16_t* __restrict__ C,
                            int N, int K) {
  int lane = threadIdx.x & 31;
  int wave = threadIdx.x >> 5;
  int row0 = blockIdx.x * 16;
  int n0   = (blockIdx.y * 4 + wave) * 64;

  v8f acc[4] = {v8f{}, v8f{}, v8f{}, v8f{}};
  const bf16_t* Arow = A + (size_t)row0 * K;
  const bf16_t* B0 = Bt + (size_t)(n0)      * K;
  const bf16_t* B1 = Bt + (size_t)(n0 + 16) * K;
  const bf16_t* B2 = Bt + (size_t)(n0 + 32) * K;
  const bf16_t* B3 = Bt + (size_t)(n0 + 48) * K;

  // prologue: k-step 0
  v16bf a_c  = load_frag(Arow, K);
  v16bf b_c0 = load_frag(B0, K);
  v16bf b_c1 = load_frag(B1, K);
  v16bf b_c2 = load_frag(B2, K);
  v16bf b_c3 = load_frag(B3, K);

  for (int k0 = 0; k0 + 32 < K; k0 += 32) {
    int kn = k0 + 32;
    // issue next k-step's loads first (overlap with WMMAs below)
    v16bf a_n  = load_frag(Arow + kn, K);
    v16bf b_n0 = load_frag(B0 + kn, K);
    v16bf b_n1 = load_frag(B1 + kn, K);
    v16bf b_n2 = load_frag(B2 + kn, K);
    v16bf b_n3 = load_frag(B3 + kn, K);

    acc[0] = wmma_bf16(a_c, b_c0, acc[0]);
    acc[1] = wmma_bf16(a_c, b_c1, acc[1]);
    acc[2] = wmma_bf16(a_c, b_c2, acc[2]);
    acc[3] = wmma_bf16(a_c, b_c3, acc[3]);

    a_c = a_n; b_c0 = b_n0; b_c1 = b_n1; b_c2 = b_n2; b_c3 = b_n3;
  }
  // epilogue k-step
  acc[0] = wmma_bf16(a_c, b_c0, acc[0]);
  acc[1] = wmma_bf16(a_c, b_c1, acc[1]);
  acc[2] = wmma_bf16(a_c, b_c2, acc[2]);
  acc[3] = wmma_bf16(a_c, b_c3, acc[3]);

  int half = lane >> 4, n = lane & 15;
#pragma unroll
  for (int t = 0; t < 4; ++t)
#pragma unroll
    for (int r = 0; r < 8; ++r)
      C[(size_t)(row0 + r + 8 * half) * N + (n0 + 16 * t + n)] =
          (bf16_t)acc[t][r];
}

// ---------------------------------------------------------------------------
// Kernel 4: RoPE + head-major transpose.
//   src: [B, S, nh*128] bf16 (projection output)
//   dst: [B, nh, S, 128] bf16, value = (x*cos + rotate_half(x)*sin) * scale
//   theta picked on-device from layer_idx (graph-capture safe).
// ---------------------------------------------------------------------------
__global__ void k_rope(const bf16_t* __restrict__ src,
                       bf16_t* __restrict__ dst,
                       int nh, float scale,
                       const int* __restrict__ layer_idx) {
  int i = blockIdx.x * blockDim.x + threadIdx.x;
  int total = B_DIM * S_DIM * nh * HEAD_DIM;
  if (i >= total) return;

  bool is_global = ((*layer_idx) % 6) == 5;
  float theta = is_global ? 1000000.0f : 10000.0f;

  int d  = i & (HEAD_DIM - 1);
  int t  = i >> 7;          // b*S*nh + s*nh + h
  int h  = t % nh;
  int t2 = t / nh;          // b*S + s
  int s  = t2 & (S_DIM - 1);
  int b  = t2 >> 10;

  float ex  = (float)(d & ~1) * (1.0f / (float)HEAD_DIM);
  float ang = (float)s * __powf(theta, -ex);
  float c   = __cosf(ang);
  float sn  = __sinf(ang);

  size_t base = (size_t)t << 7;   // index of (b,s,h,0) in src
  float x   = (float)src[base + d];
  float rot = (d < HEAD_DIM / 2) ? -(float)src[base + d + HEAD_DIM / 2]
                                 :  (float)src[base + d - HEAD_DIM / 2];
  float v = (x * c + rot * sn) * scale;

  dst[((((size_t)b * nh + h) * S_DIM) + s) * HEAD_DIM + d] = (bf16_t)v;
}

// ---------------------------------------------------------------------------
// Kernel 5: V transpose:  Vp[B,S,KV*128] -> Vt[B,KV,128,S]
// ---------------------------------------------------------------------------
__global__ void k_vt(const bf16_t* __restrict__ Vp, bf16_t* __restrict__ Vt) {
  int i = blockIdx.x * blockDim.x + threadIdx.x;
  int total = B_DIM * S_DIM * N_KV * HEAD_DIM;
  if (i >= total) return;
  int d  = i & 127;
  int kv = (i >> 7) & 3;
  int s  = (i >> 9) & (S_DIM - 1);
  int b  = i >> 19;
  Vt[(((size_t)(b * N_KV + kv)) * HEAD_DIM + d) * S_DIM + s] = Vp[i];
}

// ---------------------------------------------------------------------------
// Kernel 6: flash-style GQA attention with sliding-window-future mask.
//   Qr: [B,H,S,128] bf16 (pre-scaled by 1/sqrt(128))
//   Kr: [B,KV,S,128] bf16, Vt: [B,KV,128,S] bf16
//   Oa: [B,S,H*128] bf16
//   One wave (32 threads) per 16-query tile; grid (S/16, H, B).
//   V-fragment loads are hoisted ahead of the softmax VALU work so the
//   exp/shfl/LDS traffic hides their latency.
// ---------------------------------------------------------------------------
__global__ void k_attn(const bf16_t* __restrict__ Qr,
                       const bf16_t* __restrict__ Kr,
                       const bf16_t* __restrict__ Vt,
                       bf16_t* __restrict__ Oa,
                       const int* __restrict__ layer_idx) {
  __shared__ bf16_t P[16 * 32];   // P tile: D-layout -> A-layout shuffle

  int lane = threadIdx.x & 31;
  int half = lane >> 4;
  int n    = lane & 15;
  int q0   = blockIdx.x * 16;
  int h    = blockIdx.y;
  int b    = blockIdx.z;
  int kv   = h / GROUPS;

  bool is_global = ((*layer_idx) % 6) == 5;
  int W = is_global ? 0 : WINDOW;

  const bf16_t* Qb = Qr + (((size_t)(b * N_HEADS + h)) * S_DIM + q0) * HEAD_DIM;
  const bf16_t* Kb = Kr + ((size_t)(b * N_KV + kv)) * S_DIM * HEAD_DIM;
  const bf16_t* Vb = Vt + ((size_t)(b * N_KV + kv)) * HEAD_DIM * S_DIM;

  v16bf aQ[4];
#pragma unroll
  for (int c = 0; c < 4; ++c) aQ[c] = load_frag(Qb + c * 32, HEAD_DIM);

  v8f acc[8];
#pragma unroll
  for (int t = 0; t < 8; ++t) acc[t] = v8f{};
  float mrow[8], lrow[8];
#pragma unroll
  for (int r = 0; r < 8; ++r) { mrow[r] = -3.0e38f; lrow[r] = 0.0f; }

  int kmax = q0 + 16 + W;
  if (kmax > S_DIM) kmax = S_DIM;
  int nkt = (kmax + 31) >> 5;     // 32 keys per iteration

  for (int kt = 0; kt < nkt; ++kt) {
    int kbase = kt * 32;

    // ---- scores S = Q * K^T  (two 16-key subtiles) ----
    v8f s0 = v8f{}, s1 = v8f{};
#pragma unroll
    for (int c = 0; c < 4; ++c) {
      v16bf bK0 = load_frag(Kb + (size_t)(kbase)      * HEAD_DIM + c * 32, HEAD_DIM);
      v16bf bK1 = load_frag(Kb + (size_t)(kbase + 16) * HEAD_DIM + c * 32, HEAD_DIM);
      s0 = wmma_bf16(aQ[c], bK0, s0);
      s1 = wmma_bf16(aQ[c], bK1, s1);
    }

    // ---- hoist V fragment loads; softmax VALU below hides their latency ----
    v16bf bV[8];
#pragma unroll
    for (int t2 = 0; t2 < 8; ++t2)
      bV[t2] = load_frag(Vb + (size_t)(t2 * 16) * S_DIM + kbase, S_DIM);

    __builtin_amdgcn_wave_barrier();   // keep LDS P reuse ordered vs prior reads

    // ---- online softmax (row stats live per 16-lane half) ----
#pragma unroll
    for (int r = 0; r < 8; ++r) {
      int m  = r + 8 * half;
      int sq = q0 + m;
      float e0 = (kbase + n      <= sq + W) ? s0[r] : -3.0e38f;
      float e1 = (kbase + 16 + n <= sq + W) ? s1[r] : -3.0e38f;

      float t = fmaxf(e0, e1);
      t = fmaxf(t, __shfl_xor(t, 1, 16));
      t = fmaxf(t, __shfl_xor(t, 2, 16));
      t = fmaxf(t, __shfl_xor(t, 4, 16));
      t = fmaxf(t, __shfl_xor(t, 8, 16));
      float mnew = fmaxf(mrow[r], t);

      float p0 = __expf(e0 - mnew);
      float p1 = __expf(e1 - mnew);
      float ssum = p0 + p1;
      ssum += __shfl_xor(ssum, 1, 16);
      ssum += __shfl_xor(ssum, 2, 16);
      ssum += __shfl_xor(ssum, 4, 16);
      ssum += __shfl_xor(ssum, 8, 16);

      float alpha = __expf(mrow[r] - mnew);
      lrow[r] = lrow[r] * alpha + ssum;
      mrow[r] = mnew;

      P[m * 32 + n]      = (bf16_t)p0;
      P[m * 32 + 16 + n] = (bf16_t)p1;

#pragma unroll
      for (int t2 = 0; t2 < 8; ++t2) acc[t2][r] *= alpha;
    }

    __builtin_amdgcn_wave_barrier();
    asm volatile("s_wait_dscnt 0" ::: "memory");  // P writes visible (same wave)

    // P reloaded in A-fragment layout (ds_load_b128 x2 per lane)
    v16bf aP = frag_from(&P[(lane & 15) * 32 + half * 8]);
    __builtin_amdgcn_wave_barrier();

    // ---- O += P * V  (K=32 keys, N=128 head dims) ----
#pragma unroll
    for (int t2 = 0; t2 < 8; ++t2)
      acc[t2] = wmma_bf16(aP, bV[t2], acc[t2]);
  }

  // ---- epilogue: normalize and scatter to [B,S,H*128] ----
  bf16_t* Ob = Oa + ((size_t)b * S_DIM + q0) * (N_HEADS * HEAD_DIM) + h * HEAD_DIM;
#pragma unroll
  for (int t2 = 0; t2 < 8; ++t2)
#pragma unroll
    for (int r = 0; r < 8; ++r) {
      float v = acc[t2][r] / lrow[r];
      Ob[(size_t)(r + 8 * half) * (N_HEADS * HEAD_DIM) + t2 * 16 + n] = (bf16_t)v;
    }
}

// ---------------------------------------------------------------------------
// Kernel 7: output projection + spike (software-pipelined like k_gemm_bf16).
//   out[M,N] = heaviside(A[M,K] * WoT[N,K]^T - 1.0)  -> f32 {0,1}
// ---------------------------------------------------------------------------
__global__ void k_gemm_wo_spike(const bf16_t* __restrict__ A,
                                const bf16_t* __restrict__ Bt,
                                float* __restrict__ out,
                                int N, int K) {
  int lane = threadIdx.x & 31;
  int wave = threadIdx.x >> 5;
  int row0 = blockIdx.x * 16;
  int n0   = (blockIdx.y * 4 + wave) * 64;

  v8f acc[4] = {v8f{}, v8f{}, v8f{}, v8f{}};
  const bf16_t* Arow = A + (size_t)row0 * K;
  const bf16_t* B0 = Bt + (size_t)(n0)      * K;
  const bf16_t* B1 = Bt + (size_t)(n0 + 16) * K;
  const bf16_t* B2 = Bt + (size_t)(n0 + 32) * K;
  const bf16_t* B3 = Bt + (size_t)(n0 + 48) * K;

  v16bf a_c  = load_frag(Arow, K);
  v16bf b_c0 = load_frag(B0, K);
  v16bf b_c1 = load_frag(B1, K);
  v16bf b_c2 = load_frag(B2, K);
  v16bf b_c3 = load_frag(B3, K);

  for (int k0 = 0; k0 + 32 < K; k0 += 32) {
    int kn = k0 + 32;
    v16bf a_n  = load_frag(Arow + kn, K);
    v16bf b_n0 = load_frag(B0 + kn, K);
    v16bf b_n1 = load_frag(B1 + kn, K);
    v16bf b_n2 = load_frag(B2 + kn, K);
    v16bf b_n3 = load_frag(B3 + kn, K);

    acc[0] = wmma_bf16(a_c, b_c0, acc[0]);
    acc[1] = wmma_bf16(a_c, b_c1, acc[1]);
    acc[2] = wmma_bf16(a_c, b_c2, acc[2]);
    acc[3] = wmma_bf16(a_c, b_c3, acc[3]);

    a_c = a_n; b_c0 = b_n0; b_c1 = b_n1; b_c2 = b_n2; b_c3 = b_n3;
  }
  acc[0] = wmma_bf16(a_c, b_c0, acc[0]);
  acc[1] = wmma_bf16(a_c, b_c1, acc[1]);
  acc[2] = wmma_bf16(a_c, b_c2, acc[2]);
  acc[3] = wmma_bf16(a_c, b_c3, acc[3]);

  int half = lane >> 4, n = lane & 15;
#pragma unroll
  for (int t = 0; t < 4; ++t)
#pragma unroll
    for (int r = 0; r < 8; ++r) {
      float u = acc[t][r];
      out[(size_t)(row0 + r + 8 * half) * N + (n0 + 16 * t + n)] =
          (u - 1.0f > 0.0f) ? 1.0f : 0.0f;
    }
}

// ---------------------------------------------------------------------------
// Host launcher
// ---------------------------------------------------------------------------
extern "C" void kernel_launch(void* const* d_in, const int* in_sizes, int n_in,
                              void* d_out, int out_size, void* d_ws, size_t ws_size,
                              hipStream_t stream) {
  (void)in_sizes; (void)n_in; (void)out_size; (void)ws_size;

  const float* x  = (const float*)d_in[0];   // [B,S,2048]
  const float* Wq = (const float*)d_in[1];   // [2048,2048]
  const float* Wk = (const float*)d_in[2];   // [2048,512]
  const float* Wv = (const float*)d_in[3];   // [2048,512]
  const float* Wo = (const float*)d_in[4];   // [2048,2048]
  const int* layer_idx = (const int*)d_in[5];
  float* out = (float*)d_out;                // [B,S,2048] {0,1}

  // ---- workspace carve-up (bf16 buffers, 256B aligned) ----
  char* ws = (char*)d_ws;
  size_t off = 0;
  auto alloc = [&](size_t elems) {
    bf16_t* p = (bf16_t*)(ws + off);
    off += (elems * sizeof(bf16_t) + 255) & ~(size_t)255;
    return p;
  };
  const size_t MROWS = (size_t)B_DIM * S_DIM;                 // 2048
  bf16_t* xb  = alloc(MROWS * D_MODEL);                       // x bf16
  bf16_t* WqT = alloc((size_t)D_MODEL * D_MODEL);             // [2048,2048]
  bf16_t* WkT = alloc((size_t)(N_KV * HEAD_DIM) * D_MODEL);   // [512,2048]
  bf16_t* WvT = alloc((size_t)(N_KV * HEAD_DIM) * D_MODEL);
  bf16_t* WoT = alloc((size_t)D_MODEL * D_MODEL);
  bf16_t* Qp  = alloc(MROWS * D_MODEL);                       // q proj (pre-rope)
  bf16_t* Kp  = alloc(MROWS * (N_KV * HEAD_DIM));
  bf16_t* Vp  = alloc(MROWS * (N_KV * HEAD_DIM));
  bf16_t* Qr  = alloc(MROWS * D_MODEL);                       // [B,H,S,128]
  bf16_t* Kr  = alloc(MROWS * (N_KV * HEAD_DIM));             // [B,KV,S,128]
  bf16_t* Vt  = alloc(MROWS * (N_KV * HEAD_DIM));             // [B,KV,128,S]
  bf16_t* Oa  = Qp;  // attention output reuses Qp (free after rope)

  const int TPB = 256;

  // 1) converts / transposes
  {
    int nx = (int)(MROWS * D_MODEL);
    k_convert<<<(nx + TPB - 1) / TPB, TPB, 0, stream>>>(x, xb, nx);
    int nw = D_MODEL * D_MODEL;
    k_transpose_w<<<(nw + TPB - 1) / TPB, TPB, 0, stream>>>(Wq, WqT, D_MODEL, D_MODEL);
    k_transpose_w<<<(nw + TPB - 1) / TPB, TPB, 0, stream>>>(Wo, WoT, D_MODEL, D_MODEL);
    int nkv = D_MODEL * N_KV * HEAD_DIM;
    k_transpose_w<<<(nkv + TPB - 1) / TPB, TPB, 0, stream>>>(Wk, WkT, D_MODEL, N_KV * HEAD_DIM);
    k_transpose_w<<<(nkv + TPB - 1) / TPB, TPB, 0, stream>>>(Wv, WvT, D_MODEL, N_KV * HEAD_DIM);
  }

  // 2) QKV projections (bf16 WMMA GEMMs)
  {
    dim3 gq((unsigned)(MROWS / 16), D_MODEL / 256);
    k_gemm_bf16<<<gq, 128, 0, stream>>>(xb, WqT, Qp, D_MODEL, D_MODEL);
    dim3 gk((unsigned)(MROWS / 16), (N_KV * HEAD_DIM) / 256);
    k_gemm_bf16<<<gk, 128, 0, stream>>>(xb, WkT, Kp, N_KV * HEAD_DIM, D_MODEL);
    k_gemm_bf16<<<gk, 128, 0, stream>>>(xb, WvT, Vp, N_KV * HEAD_DIM, D_MODEL);
  }

  // 3) RoPE (+softmax scale on Q) and layout transposes
  {
    int nq = (int)(MROWS * D_MODEL);
    k_rope<<<(nq + TPB - 1) / TPB, TPB, 0, stream>>>(Qp, Qr, N_HEADS, SM_SCALE, layer_idx);
    int nk = (int)(MROWS * N_KV * HEAD_DIM);
    k_rope<<<(nk + TPB - 1) / TPB, TPB, 0, stream>>>(Kp, Kr, N_KV, 1.0f, layer_idx);
    k_vt<<<(nk + TPB - 1) / TPB, TPB, 0, stream>>>(Vp, Vt);
  }

  // 4) flash attention (one wave per 16-query tile)
  {
    dim3 ga(S_DIM / 16, N_HEADS, B_DIM);
    k_attn<<<ga, 32, 0, stream>>>(Qr, Kr, Vt, Oa, layer_idx);
  }

  // 5) output projection + spike -> d_out (f32)
  {
    dim3 go((unsigned)(MROWS / 16), D_MODEL / 256);
    k_gemm_wo_spike<<<go, 128, 0, stream>>>(Oa, WoT, out, D_MODEL, D_MODEL);
  }
}
